// TinyLLM_62466004353690
// MI455X (gfx1250) — matrix operand order repaired and checked by reading, not compiled, not run on hardware
//
#include <hip/hip_runtime.h>
#include <hip/hip_bf16.h>
#include <cstdint>

// ---------------------------------------------------------------------------
// MI455X (gfx1250) GPT-2 tiny forward.
// Compute-bound (~740 GFLOP vs ~0.5 GB HBM traffic): all GEMM-like work on
// v_wmma_f32_16x16x32_bf16 (fp32 accumulate). GEMM is software-pipelined:
// the Tensor Data Mover DMAs the NEXT A tile (double-buffered LDS) while
// WMMAs consume the current one (s_wait_tensorcnt 1 keeps one DMA in
// flight); the W tile for the next step is loaded to registers before the
// WMMA block and stored (f32->bf16, transposed, packed b32) after it.
// ---------------------------------------------------------------------------

typedef __bf16 v16bf __attribute__((ext_vector_type(16)));
typedef float  v8f   __attribute__((ext_vector_type(8)));
typedef unsigned int v4u __attribute__((ext_vector_type(4)));
typedef int          v8i __attribute__((ext_vector_type(8)));
typedef int          v4i __attribute__((ext_vector_type(4)));

#if defined(__has_builtin)
#  if __has_builtin(__builtin_amdgcn_tensor_load_to_lds)
#    define USE_TDM 1
#  endif
#endif
#ifndef USE_TDM
#  define USE_TDM 0
#endif

__device__ __forceinline__ uint16_t f2bf(float f) {
    uint32_t u = __float_as_uint(f);
    u += 0x7FFFu + ((u >> 16) & 1u);            // round-to-nearest-even
    return (uint16_t)(u >> 16);
}
__device__ __forceinline__ uint32_t pack_bf2(float lo, float hi) {
    return (uint32_t)f2bf(lo) | ((uint32_t)f2bf(hi) << 16);
}

union AFrag { v16bf v; uint4 q[2]; };

__device__ __forceinline__ v8f wmma_bf16(const v16bf& a, const v16bf& b, v8f c) {
    return __builtin_amdgcn_wmma_f32_16x16x32_bf16(false, a, false, b,
                                                   (short)0, c, false, false);
}

#if USE_TDM
// ---------------------------------------------------------------------------
// TDM: DMA a 2-D tile of bf16 (tile_w x tile_h elems, row stride
// stride_elems) from global to LDS. g1w0 carries data_size + LDS pad config
// (pad keeps the same +16B/row LDS padding the compute side indexes with).
// 6-arg builtin form (clang-23 / therock-10.0): trailing int32x8 is the
// unused second descriptor slot (zero for <=2-D tensors).
// ---------------------------------------------------------------------------
__device__ __forceinline__ void tdm_load_2d(uint32_t lds_addr, const void* gptr,
                                            uint32_t dim0, uint32_t dim1,
                                            uint32_t stride_elems,
                                            uint32_t tile_w, uint32_t tile_h,
                                            uint32_t g1w0) {
    const uint64_t ga = (uint64_t)(uintptr_t)gptr;
    const v4u g0 = {
        1u,                                                   // count=1, user
        lds_addr,                                             // lds_addr
        (uint32_t)ga,                                         // global_addr lo
        ((uint32_t)(ga >> 32) & 0x1FFFFFFu) | (2u << 30)      // addr hi | type=2
    };
    const v8i g1 = {
        (int)g1w0,                                            // mask=0|dsize|pad
        (int)((dim0 & 0xFFFFu) << 16),                        // abar=0 | dim0.lo
        (int)((dim0 >> 16) | ((dim1 & 0xFFFFu) << 16)),       // dim0.hi | dim1.lo
        (int)((dim1 >> 16) | (tile_w << 16)),                 // dim1.hi | tile0
        (int)tile_h,                                          // tile1 | tile2=0
        (int)stride_elems,                                    // dim0_stride lo32
        0,                                                    // stride hi | d1s.lo
        0                                                     // d1s.hi
    };
    const v4i z4 = {0, 0, 0, 0};
    const v8i z8 = {0, 0, 0, 0, 0, 0, 0, 0};
    __builtin_amdgcn_tensor_load_to_lds(g0, g1, z4, z4, z8, 0);
}
// data_size=2B | pad_enable | pad_interval | pad_amount(4 DWORDs = 16B)
#define TDM_W0_ROW64B  (0x10000u | (1u << 20) | (3u << 22) | (3u << 25))  // 32 bf16/row
#define TDM_W0_ROW128B (0x10000u | (1u << 20) | (4u << 22) | (3u << 25))  // 64 bf16/row
__device__ __forceinline__ uint32_t lds_off(const void* p) {
    return (uint32_t)(uintptr_t)p;     // generic LDS addr: low 32 bits = offset
}
#endif

// ---------------------------------------------------------------------------
// Embedding: x[m][e] = tok_emb[idx[m]][e] + pos_emb[m % T][e]
// ---------------------------------------------------------------------------
__global__ __launch_bounds__(256)
void embed_kernel(const int* __restrict__ idx, const float* __restrict__ tok,
                  const float* __restrict__ pos, float* __restrict__ x,
                  int T, int E) {
    const int m  = blockIdx.x;
    const int t  = m % T;
    const int id = idx[m];
    for (int e = threadIdx.x; e < E; e += 256)
        x[(size_t)m * E + e] = tok[(size_t)id * E + e] + pos[(size_t)t * E + e];
}

// ---------------------------------------------------------------------------
// LayerNorm (fp32 in) -> bf16 out (WMMA A operand)
// ---------------------------------------------------------------------------
__global__ __launch_bounds__(256)
void layernorm_bf16(const float* __restrict__ x, const float* __restrict__ g,
                    const float* __restrict__ b, uint16_t* __restrict__ out,
                    int E) {
    const int row = blockIdx.x;
    const int tid = threadIdx.x;
    const float* xr = x + (size_t)row * E;

    float s = 0.f, ss = 0.f;
    for (int e = tid; e < E; e += 256) { float v = xr[e]; s += v; ss += v * v; }

    __shared__ float r1[256];
    __shared__ float r2[256];
    r1[tid] = s; r2[tid] = ss;
    __syncthreads();
    for (int off = 128; off; off >>= 1) {
        if (tid < off) { r1[tid] += r1[tid + off]; r2[tid] += r2[tid + off]; }
        __syncthreads();
    }
    const float mean = r1[0] / (float)E;
    const float var  = r2[0] / (float)E - mean * mean;
    const float inv  = rsqrtf(var + 1e-5f);
    for (int e = tid; e < E; e += 256)
        out[(size_t)row * E + e] = f2bf((xr[e] - mean) * inv * g[e] + b[e]);
}

// ---------------------------------------------------------------------------
// Tiled WMMA GEMM: C[M,N] = A_bf16[M,K] @ W_f32[K,N]
//   MODE 0: store f32        MODE 1: C_f32 += result (fused residual)
//   MODE 2: store bf16       MODE 3: exact GELU -> bf16
// Block tile 128x128x32, 8 waves (2M x 4N), wave tile 64x32 = 8 WMMA/k-step.
// Double-buffered LDS; A tiles DMA'd by TDM one step ahead; W tile for the
// next step pipelined through registers around the WMMA block.
// ---------------------------------------------------------------------------
#define BM 128
#define BN 128
#define BK 32

template <int MODE>
__global__ __launch_bounds__(256)
void gemm_bf16(const uint16_t* __restrict__ A, const float* __restrict__ W,
               void* __restrict__ Cout, int M, int N, int K) {
    __shared__ __align__(16) uint16_t As[2][BM][BK + 8];
    __shared__ __align__(16) uint16_t Bt[2][BN][BK + 8];  // n-major (transposed)

    const int tid  = threadIdx.x;
    const int lane = tid & 31;
    const int wave = tid >> 5;
    const int wm   = wave & 1;        // 0..1  -> 64-row half
    const int wn   = wave >> 1;       // 0..3  -> 32-col quarter
    const int l16  = lane & 15;
    const int half = lane >> 4;
    const int n0   = blockIdx.x * BN;
    const int m0   = blockIdx.y * BM;
    const int nk   = K / BK;

    // per-thread W-staging coordinates (2 chunks x 4 cols x 2 k-rows)
    const int kp0 = tid >> 5;                 // chunk0 k-pair 0..7
    const int nc0 = (tid & 31) * 4;           // cols 0..124

    v8f acc[4][2];
    const v8f z8 = {0.f,0.f,0.f,0.f,0.f,0.f,0.f,0.f};
#pragma unroll
    for (int mi = 0; mi < 4; ++mi)
#pragma unroll
        for (int ni = 0; ni < 2; ++ni) acc[mi][ni] = z8;

    // ---- prologue: tile 0 ----
#if USE_TDM
    if (wave == 0)
        tdm_load_2d(lds_off(&As[0][0][0]), &A[(size_t)m0 * K],
                    (uint32_t)K, (uint32_t)M, (uint32_t)K,
                    BK, BM, TDM_W0_ROW64B);
#else
#pragma unroll
    for (int i = 0; i < 2; ++i) {
        int c = tid + i * 256;
        int r = c >> 2, cc = (c & 3) * 8;
        *(uint4*)&As[0][r][cc] = *(const uint4*)&A[(size_t)(m0 + r) * K + cc];
    }
#endif
#pragma unroll
    for (int i = 0; i < 2; ++i) {             // stage W tile 0 directly
        const int kp = kp0 + i * 8;
        const int kr = kp * 2;
#pragma unroll
        for (int j = 0; j < 4; ++j) {
            int n = n0 + nc0 + j;
            float w0 = 0.f, w1 = 0.f;
            if (n < N) {
                w0 = W[(size_t)kr * N + n];
                w1 = W[(size_t)(kr + 1) * N + n];
            }
            *(uint32_t*)&Bt[0][nc0 + j][kp * 2] = pack_bf2(w0, w1);
        }
    }

    for (int kt = 0; kt < nk; ++kt) {
        const int cur = kt & 1, nxt = cur ^ 1;
        const bool has_next = (kt + 1 < nk);
        const int knext = (kt + 1) * BK;

#if USE_TDM
        // DMA next A tile into the spare buffer while we compute this one.
        if (has_next && wave == 0)
            tdm_load_2d(lds_off(&As[nxt][0][0]), &A[(size_t)m0 * K + knext],
                        (uint32_t)K, (uint32_t)M, (uint32_t)K,
                        BK, BM, TDM_W0_ROW64B);
        // TENSORcnt is in-order: <=1 pending means the CURRENT tile landed.
        // (builtin needs a constant immediate -> uniform two-way branch)
        if (wave == 0) {
            if (has_next) __builtin_amdgcn_s_wait_tensorcnt(1);
            else          __builtin_amdgcn_s_wait_tensorcnt(0);
        }
#else
        // fallback: stage current A tile synchronously
#pragma unroll
        for (int i = 0; i < 2; ++i) {
            int c = tid + i * 256;
            int r = c >> 2, cc = (c & 3) * 8;
            *(uint4*)&As[cur][r][cc] =
                *(const uint4*)&A[(size_t)(m0 + r) * K + kt * BK + cc];
        }
#endif
        __syncthreads();

        // ---- issue next W tile's global loads (hidden under WMMA) ----
        float wr[2][4][2];
        if (has_next) {
#pragma unroll
            for (int i = 0; i < 2; ++i) {
                const int kr = knext + (kp0 + i * 8) * 2;
#pragma unroll
                for (int j = 0; j < 4; ++j) {
                    int n = n0 + nc0 + j;
                    wr[i][j][0] = (n < N) ? W[(size_t)kr * N + n] : 0.f;
                    wr[i][j][1] = (n < N) ? W[(size_t)(kr + 1) * N + n] : 0.f;
                }
            }
            if (kt + 2 < nk && n0 < N)
                __builtin_prefetch(
                    &W[(size_t)((kt + 2) * BK + (tid >> 3)) * N + n0], 0, 0);
        }

        // ---- compute tile kt ----
        AFrag afr[4], bfr[2];
#pragma unroll
        for (int mi = 0; mi < 4; ++mi) {
            const uint16_t* p = &As[cur][wm * 64 + mi * 16 + l16][half * 16];
            afr[mi].q[0] = *(const uint4*)p;
            afr[mi].q[1] = *(const uint4*)(p + 8);
        }
#pragma unroll
        for (int ni = 0; ni < 2; ++ni) {
            const uint16_t* p = &Bt[cur][wn * 32 + ni * 16 + l16][half * 16];
            bfr[ni].q[0] = *(const uint4*)p;
            bfr[ni].q[1] = *(const uint4*)(p + 8);
        }
#pragma unroll
        for (int mi = 0; mi < 4; ++mi)
#pragma unroll
            for (int ni = 0; ni < 2; ++ni)
                acc[mi][ni] = wmma_bf16(afr[mi].v, bfr[ni].v, acc[mi][ni]);

        // ---- store next W tile (convert + transpose, packed b32) ----
        if (has_next) {
#pragma unroll
            for (int i = 0; i < 2; ++i) {
                const int kp = kp0 + i * 8;
#pragma unroll
                for (int j = 0; j < 4; ++j)
                    *(uint32_t*)&Bt[nxt][nc0 + j][kp * 2] =
                        pack_bf2(wr[i][j][0], wr[i][j][1]);
            }
        }
        __syncthreads();
    }

    // ---- epilogue (C layout: VGPR vi -> rows vi / vi+8; cols across lanes) --
#pragma unroll
    for (int mi = 0; mi < 4; ++mi)
#pragma unroll
        for (int ni = 0; ni < 2; ++ni) {
            const int col = n0 + wn * 32 + ni * 16 + l16;
            if (col >= N) continue;
#pragma unroll
            for (int vi = 0; vi < 8; ++vi) {
                const int row = m0 + wm * 64 + mi * 16 + half * 8 + vi;
                const float val = acc[mi][ni][vi];
                const size_t o = (size_t)row * N + col;
                if (MODE == 0)      ((float*)Cout)[o]  = val;
                else if (MODE == 1) ((float*)Cout)[o] += val;
                else if (MODE == 2) ((uint16_t*)Cout)[o] = f2bf(val);
                else {
                    float gv = 0.5f * val * (1.f + erff(val * 0.70710678f));
                    ((uint16_t*)Cout)[o] = f2bf(gv);
                }
            }
        }
}

// ---------------------------------------------------------------------------
// Causal flash attention, head_dim = 64.
// Grid: (T/128, B*H). 8 waves, each wave owns 16 query rows x all 64 keys of
// the current key block. QK^T and P@V both on WMMA. Q/K tiles via TDM; V
// transposed manually (packed 32-bit LDS stores). P transposed through the
// wave's private LDS region (same-wave LDS ops are in-order).
// ---------------------------------------------------------------------------
__global__ __launch_bounds__(256)
void attention_kernel(const uint16_t* __restrict__ q,
                      const uint16_t* __restrict__ k,
                      const uint16_t* __restrict__ v,
                      uint16_t* __restrict__ att, int T, int Hn) {
    const int E  = Hn * 64;
    const int qb = blockIdx.x;
    const int bh = blockIdx.y;
    const int b  = bh / Hn, h = bh % Hn;
    const int q0 = qb * 128;

    __shared__ __align__(16) uint16_t Qs[128][72];
    __shared__ __align__(16) uint16_t Ks[64][72];
    __shared__ __align__(16) uint16_t Vt[64][72];   // [d][t]
    __shared__ __align__(16) uint16_t Ps[128][72];  // per-wave 16-row slices

    const int tid  = threadIdx.x;
    const int lane = tid & 31;
    const int wave = tid >> 5;
    const int l16  = lane & 15;
    const int half = lane >> 4;

    // stage Q tile 128x64
#if USE_TDM
    if (wave == 0)
        tdm_load_2d(lds_off(&Qs[0][0]),
                    &q[((size_t)(b * T + q0)) * E + h * 64],
                    (uint32_t)E, 128u, (uint32_t)E, 64u, 128u, TDM_W0_ROW128B);
#else
#pragma unroll
    for (int i = 0; i < 4; ++i) {
        int c = tid + i * 256;                     // 0..1023
        int t = c >> 3, d8 = (c & 7) * 8;
        *(uint4*)&Qs[t][d8] =
            *(const uint4*)&q[((size_t)(b * T + q0 + t)) * E + h * 64 + d8];
    }
#endif

    const v8f z8 = {0.f,0.f,0.f,0.f,0.f,0.f,0.f,0.f};
    v8f o[4];
#pragma unroll
    for (int di = 0; di < 4; ++di) o[di] = z8;
    float mrow[8], lrow[8];
#pragma unroll
    for (int vi = 0; vi < 8; ++vi) { mrow[vi] = -INFINITY; lrow[vi] = 0.f; }

    const int nkb = 2 * qb + 2;                    // causal: key blocks 0..nkb-1
    for (int j = 0; j < nkb; ++j) {
        const int t0 = j * 64;
        __syncthreads();                           // prev-iter reads done
        // ---- stage K (TDM) and V transposed (packed stores): 64x64 each ----
#if USE_TDM
        if (wave == 0)
            tdm_load_2d(lds_off(&Ks[0][0]),
                        &k[((size_t)(b * T + t0)) * E + h * 64],
                        (uint32_t)E, 64u, (uint32_t)E, 64u, 64u, TDM_W0_ROW128B);
#else
#pragma unroll
        for (int i = 0; i < 2; ++i) {
            int c = tid + i * 256;
            int t = c >> 3, d8 = (c & 7) * 8;
            *(uint4*)&Ks[t][d8] =
                *(const uint4*)&k[((size_t)(b * T + t0 + t)) * E + h * 64 + d8];
        }
#endif
        {
            int t2 = (tid >> 3) * 2;               // key-row pair 0..62
            int d8 = (tid & 7) * 8;
            const size_t base =
                ((size_t)(b * T + t0 + t2)) * E + h * 64 + d8;
            uint4 v0 = *(const uint4*)&v[base];
            uint4 v1 = *(const uint4*)&v[base + E];
            const uint16_t* p0 = (const uint16_t*)&v0;
            const uint16_t* p1 = (const uint16_t*)&v1;
#pragma unroll
            for (int jj = 0; jj < 8; ++jj)
                *(uint32_t*)&Vt[d8 + jj][t2] =
                    (uint32_t)p0[jj] | ((uint32_t)p1[jj] << 16);
        }
#if USE_TDM
        if (wave == 0) __builtin_amdgcn_s_wait_tensorcnt(0);
#endif
        __syncthreads();

        // ---- S = Q K^T : wave rows [wave*16, +16), cols 0..63 ----
        v8f s[4];
#pragma unroll
        for (int ni = 0; ni < 4; ++ni) s[ni] = z8;
#pragma unroll
        for (int ks = 0; ks < 2; ++ks) {
            AFrag af;
            const uint16_t* pa = &Qs[wave * 16 + l16][ks * 32 + half * 16];
            af.q[0] = *(const uint4*)pa;
            af.q[1] = *(const uint4*)(pa + 8);
#pragma unroll
            for (int ni = 0; ni < 4; ++ni) {
                AFrag bf;
                const uint16_t* pb = &Ks[ni * 16 + l16][ks * 32 + half * 16];
                bf.q[0] = *(const uint4*)pb;
                bf.q[1] = *(const uint4*)(pb + 8);
                s[ni] = wmma_bf16(af.v, bf.v, s[ni]);
            }
        }

        // ---- online softmax (row stats over 16-lane groups) ----
#pragma unroll
        for (int vi = 0; vi < 8; ++vi) {
            const int r = q0 + wave * 16 + half * 8 + vi;   // global query row
            float mnew = -INFINITY;
#pragma unroll
            for (int ni = 0; ni < 4; ++ni) {
                float sv = s[ni][vi] * 0.125f;              // 1/sqrt(64)
                const int tg = t0 + ni * 16 + l16;
                if (tg > r) sv = -INFINITY;                 // causal mask
                s[ni][vi] = sv;
                mnew = fmaxf(mnew, sv);
            }
#pragma unroll
            for (int off = 1; off < 16; off <<= 1)
                mnew = fmaxf(mnew, __shfl_xor(mnew, off, 32));

            const float mold = mrow[vi];
            const float mnow = fmaxf(mold, mnew);
            const float resc = (mold == -INFINITY) ? 0.f : __expf(mold - mnow);

            float psum = 0.f;
#pragma unroll
            for (int ni = 0; ni < 4; ++ni) {
                float p = __expf(s[ni][vi] - mnow);
                s[ni][vi] = p;
                psum += p;
            }
#pragma unroll
            for (int off = 1; off < 16; off <<= 1)
                psum += __shfl_xor(psum, off, 32);

            lrow[vi] = lrow[vi] * resc + psum;
            mrow[vi] = mnow;
#pragma unroll
            for (int di = 0; di < 4; ++di) o[di][vi] *= resc;

            // write P row (wave-private LDS region -> no barrier needed)
            const int rl = wave * 16 + half * 8 + vi;
#pragma unroll
            for (int ni = 0; ni < 4; ++ni)
                Ps[rl][ni * 16 + l16] = f2bf(s[ni][vi]);
        }

        // ---- O += P (16x64) @ V (64x64) ----
#pragma unroll
        for (int ks = 0; ks < 2; ++ks) {
            AFrag af;
            const uint16_t* pa = &Ps[wave * 16 + l16][ks * 32 + half * 16];
            af.q[0] = *(const uint4*)pa;
            af.q[1] = *(const uint4*)(pa + 8);
#pragma unroll
            for (int di = 0; di < 4; ++di) {
                AFrag bf;
                const uint16_t* pb = &Vt[di * 16 + l16][ks * 32 + half * 16];
                bf.q[0] = *(const uint4*)pb;
                bf.q[1] = *(const uint4*)(pb + 8);
                o[di] = wmma_bf16(af.v, bf.v, o[di]);
            }
        }
    }

    // ---- write out, layout [B*T][E] with head column offset ----
#pragma unroll
    for (int vi = 0; vi < 8; ++vi) {
        const float inv = 1.f / lrow[vi];
        const int r = q0 + wave * 16 + half * 8 + vi;
#pragma unroll
        for (int di = 0; di < 4; ++di)
            att[((size_t)(b * T + r)) * E + h * 64 + di * 16 + l16] =
                f2bf(o[di][vi] * inv);
    }
}

// ---------------------------------------------------------------------------
// Host orchestration
// ---------------------------------------------------------------------------
extern "C" void kernel_launch(void* const* d_in, const int* in_sizes, int n_in,
                              void* d_out, int out_size, void* d_ws, size_t ws_size,
                              hipStream_t stream) {
    (void)in_sizes; (void)n_in; (void)out_size; (void)ws_size;
    constexpr int Lc = 6, Hn = 12, E = 768, Vv = 50257, Tc = 1024, Bc = 4;
    constexpr int M = Bc * Tc;                      // 4096
    constexpr int E4 = 4 * E;                       // 3072

    const int*   idx   = (const int*)  d_in[0];
    const float* tok   = (const float*)d_in[1];
    const float* pos   = (const float*)d_in[2];
    const float* Wq    = (const float*)d_in[3];
    const float* Wk    = (const float*)d_in[4];
    const float* Wv    = (const float*)d_in[5];
    const float* Wo    = (const float*)d_in[6];
    const float* W1    = (const float*)d_in[7];
    const float* W2    = (const float*)d_in[8];
    const float* ln1g  = (const float*)d_in[9];
    const float* ln1b  = (const float*)d_in[10];
    const float* ln2g  = (const float*)d_in[11];
    const float* ln2b  = (const float*)d_in[12];
    const float* lnfg  = (const float*)d_in[13];
    const float* lnfb  = (const float*)d_in[14];
    const float* Whead = (const float*)d_in[15];
    float* logits = (float*)d_out;

    // workspace carve (all sizes 256B-aligned)
    char* w = (char*)d_ws;
    float*    x   = (float*)w;    w += (size_t)M * E * 4;
    uint16_t* h   = (uint16_t*)w; w += (size_t)M * E * 2;
    uint16_t* qb  = (uint16_t*)w; w += (size_t)M * E * 2;
    uint16_t* kb  = (uint16_t*)w; w += (size_t)M * E * 2;
    uint16_t* vb  = (uint16_t*)w; w += (size_t)M * E * 2;
    uint16_t* att = (uint16_t*)w; w += (size_t)M * E * 2;
    uint16_t* g1  = (uint16_t*)w; w += (size_t)M * E4 * 2;

    const dim3 blk(256);
    const dim3 gEE(E / BN, M / BM);                 // 6 x 32
    const dim3 gE4(E4 / BN, M / BM);                // 24 x 32
    const dim3 gHd((Vv + BN - 1) / BN, M / BM);     // 393 x 32
    const dim3 gAtt(Tc / 128, Bc * Hn);             // 8 x 48

    embed_kernel<<<M, blk, 0, stream>>>(idx, tok, pos, x, Tc, E);

    for (int l = 0; l < Lc; ++l) {
        const size_t oEE = (size_t)l * E * E;
        const size_t oE4 = (size_t)l * E * E4;

        layernorm_bf16<<<M, blk, 0, stream>>>(x, ln1g + l * E, ln1b + l * E, h, E);
        gemm_bf16<2><<<gEE, blk, 0, stream>>>(h, Wq + oEE, qb, M, E, E);
        gemm_bf16<2><<<gEE, blk, 0, stream>>>(h, Wk + oEE, kb, M, E, E);
        gemm_bf16<2><<<gEE, blk, 0, stream>>>(h, Wv + oEE, vb, M, E, E);

        attention_kernel<<<gAtt, blk, 0, stream>>>(qb, kb, vb, att, Tc, Hn);

        gemm_bf16<1><<<gEE, blk, 0, stream>>>(att, Wo + oEE, x, M, E, E);

        layernorm_bf16<<<M, blk, 0, stream>>>(x, ln2g + l * E, ln2b + l * E, h, E);
        gemm_bf16<3><<<gE4, blk, 0, stream>>>(h, W1 + oE4, g1, M, E4, E);
        gemm_bf16<1><<<gEE, blk, 0, stream>>>(g1, W2 + oE4, x, M, E, E4);
    }

    layernorm_bf16<<<M, blk, 0, stream>>>(x, lnfg, lnfb, h, E);
    gemm_bf16<0><<<gHd, blk, 0, stream>>>(h, Whead, logits, M, Vv, E);
}
